// YemongTemporal_807453851687
// MI455X (gfx1250) — compile-verified, matrix operand order checked by reading, not run on hardware
//
#include <hip/hip_runtime.h>
#include <hip/hip_bf16.h>
#include <cstdint>
#include <cstddef>

typedef __attribute__((ext_vector_type(16))) _Float16 v16h;
typedef __attribute__((ext_vector_type(8)))  _Float16 v8h;
typedef __attribute__((ext_vector_type(8)))  float    v8f;

#define BB   16
#define TT   1024
#define DM   512
#define DI   1024
#define DSZ  16
#define DTRK 32
#define NL   8
#define MTOK (BB*TT)        // 16384 tokens

enum { EPI_NONE = 0, EPI_BIAS = 1, EPI_BIAS_SOFTPLUS = 2, EPI_ACC = 3 };

// ---------------------------------------------------------------------------
// device helpers
// ---------------------------------------------------------------------------
__device__ __forceinline__ float sigmoidf_(float v) {
    return 1.0f / (1.0f + __expf(-v));
}
__device__ __forceinline__ float siluf_(float v) {
    return v * sigmoidf_(v);
}

// Load a 16x32 (f16) WMMA A/B fragment for this lane.
// Per CDNA5 ISA: lanes 0-15 hold K = [kk..kk+7] and [kk+16..kk+23],
// lanes 16-31 hold K = [kk+8..kk+15] and [kk+24..kk+31].
// `p` must point at rowStart + kk + koff, where koff = (lane>>4)*8.
__device__ __forceinline__ v16h load_frag16(const _Float16* __restrict__ p) {
    v8h lo = *reinterpret_cast<const v8h*>(p);
    v8h hi = *reinterpret_cast<const v8h*>(p + 16);
    v16h r;
#pragma unroll
    for (int i = 0; i < 8; ++i) { r[i] = lo[i]; r[i + 8] = hi[i]; }
    return r;
}

// ---------------------------------------------------------------------------
// WMMA GEMM:  C[M,N](f32) = A[M,K](f16) * W[N,K](f16)^T  (+bias / +softplus / +=)
// block = 256 threads = 8 waves arranged WM x WN; wave tile = (MT*16) x (NT*16).
// Block tile = (WM*MT*16) x (WN*NT*16).  Requires M % (WM*MT*16) == 0,
// N % 16 == 0, K % 32 == 0.  All tile-validity branches are wave-uniform so
// EXEC is all-ones at every WMMA.
// ---------------------------------------------------------------------------
template<int MT, int NT, int WM, int WN>
__global__ __launch_bounds__(256)
void gemm_f16_wmma(const _Float16* __restrict__ A, const _Float16* __restrict__ W,
                   float* __restrict__ C, int M, int N, int K,
                   const float* __restrict__ bias, int mode) {
    static_assert(WM * WN == 8, "8 waves per block");
    const int lane = threadIdx.x & 31;
    const int wv   = threadIdx.x >> 5;
    const int wm   = wv / WN;
    const int wn   = wv % WN;

    const int mBase = blockIdx.x * (WM * MT * 16) + wm * (MT * 16);
    const int nBase = blockIdx.y * (WN * NT * 16) + wn * (NT * 16);
    if (nBase >= N) return;                       // wave-uniform

    const int r    = lane & 15;
    const int koff = (lane >> 4) * 8;

    const _Float16* aRow[MT];
#pragma unroll
    for (int i = 0; i < MT; ++i)
        aRow[i] = A + (size_t)(mBase + i * 16 + r) * K + koff;

    const _Float16* bRow[NT];
#pragma unroll
    for (int j = 0; j < NT; ++j) {
        const int nb = nBase + j * 16;
        // clamp invalid tiles onto tile 0's rows (results discarded at store)
        bRow[j] = W + (size_t)(((nb + 16 <= N) ? nb : nBase) + r) * K + koff;
    }

    v8f acc[MT][NT] = {};

    for (int kk = 0; kk < K; kk += 32) {
        v16h a[MT];
        v16h b[NT];
#pragma unroll
        for (int i = 0; i < MT; ++i) a[i] = load_frag16(aRow[i] + kk);
#pragma unroll
        for (int j = 0; j < NT; ++j) b[j] = load_frag16(bRow[j] + kk);
        if (kk + 32 < K) {
#pragma unroll
            for (int i = 0; i < MT; ++i) __builtin_prefetch(aRow[i] + kk + 32, 0, 1);
        }
#pragma unroll
        for (int i = 0; i < MT; ++i)
#pragma unroll
            for (int j = 0; j < NT; ++j)
                acc[i][j] = __builtin_amdgcn_wmma_f32_16x16x32_f16(
                    false, a[i], false, b[j], (short)0, acc[i][j], false, false);
    }

    // C/D layout: VGPR e -> row (e + 8*(lane>>4)), col = lane&15
    const int rowOff = (lane >> 4) * 8;
#pragma unroll
    for (int j = 0; j < NT; ++j) {
        if (nBase + (j + 1) * 16 > N) break;      // wave-uniform
        const int col = nBase + j * 16 + (lane & 15);
        const float bv = (mode == EPI_BIAS || mode == EPI_BIAS_SOFTPLUS) ? bias[col] : 0.0f;
#pragma unroll
        for (int i = 0; i < MT; ++i) {
#pragma unroll
            for (int e = 0; e < 8; ++e) {
                const int row = mBase + i * 16 + rowOff + e;
                float v = acc[i][j][e];
                if (mode == EPI_BIAS) {
                    v += bv;
                } else if (mode == EPI_BIAS_SOFTPLUS) {
                    v += bv;
                    v = (v > 20.0f) ? v : __logf(1.0f + __expf(v));
                }
                const size_t idx = (size_t)row * (size_t)N + (size_t)col;
                if (mode == EPI_ACC) C[idx] += v;
                else                 C[idx]  = v;
            }
        }
    }
}

// ---------------------------------------------------------------------------
// generic f32 -> f16 convert with row-stride + K padding
// dst[r*dK + k] = (k < min(sK,dK)) ? src[r*sK + k] : 0
// ---------------------------------------------------------------------------
__global__ __launch_bounds__(256)
void f32_to_f16_pad(const float* __restrict__ src, _Float16* __restrict__ dst,
                    long total, int sK, int dK) {
    long idx = (long)blockIdx.x * blockDim.x + threadIdx.x;
    if (idx >= total) return;
    int k = (int)(idx % dK);
    long row = idx / dK;
    float v = (k < sK) ? src[row * (long)sK + k] : 0.0f;
    dst[idx] = (_Float16)v;
}

// ---------------------------------------------------------------------------
// front end: s_emb = silu(state @ state_w^T + b), a_emb = silu(pa @ act_w^T + b)
// writes concat [s_emb, a_emb] as f16, row stride 640.  (f32 weights, tiny K)
// ---------------------------------------------------------------------------
__global__ __launch_bounds__(256)
void embed_kernel(const float* __restrict__ state, const float* __restrict__ pa,
                  const float* __restrict__ sw, const float* __restrict__ sb,
                  const float* __restrict__ aw, const float* __restrict__ ab,
                  _Float16* __restrict__ out) {
    long idx = (long)blockIdx.x * blockDim.x + threadIdx.x;
    const long total = (long)MTOK * (DM + 128);
    if (idx >= total) return;
    int  j  = (int)(idx % (DM + 128));
    long bt = idx / (DM + 128);
    float acc;
    if (j < DM) {
        acc = sb[j];
        const float* srow = state + bt * 32;
        const float* wrow = sw + (size_t)j * 32;
#pragma unroll
        for (int k = 0; k < 32; ++k) acc += srow[k] * wrow[k];
    } else {
        int jj = j - DM;
        acc = ab[jj];
        const float* prow = pa + bt * 8;
        const float* wrow = aw + (size_t)jj * 8;
#pragma unroll
        for (int k = 0; k < 8; ++k) acc += prow[k] * wrow[k];
    }
    out[idx] = (_Float16)siluf_(acc);
}

// ---------------------------------------------------------------------------
// rmsnorm helpers (one 256-thread block per token, D = 512)
// ---------------------------------------------------------------------------
__device__ __forceinline__ float block_sumsq(const float* __restrict__ row, int D) {
    float ss = 0.0f;
    for (int i = threadIdx.x; i < D; i += blockDim.x) { float v = row[i]; ss += v * v; }
#pragma unroll
    for (int off = 16; off > 0; off >>= 1) ss += __shfl_down(ss, off);
    __shared__ float red[8];
    if ((threadIdx.x & 31) == 0) red[threadIdx.x >> 5] = ss;
    __syncthreads();
    float tot = 0.0f;
#pragma unroll
    for (int w = 0; w < 8; ++w) tot += red[w];
    return tot;
}

__global__ __launch_bounds__(256)
void rmsnorm_to_f16(const float* __restrict__ x, const float* __restrict__ g,
                    _Float16* __restrict__ out, int D) {
    const float* row = x + (size_t)blockIdx.x * D;
    float tot = block_sumsq(row, D);
    float scale = rsqrtf(tot / (float)D + 1e-5f);
    _Float16* orow = out + (size_t)blockIdx.x * D;
    for (int i = threadIdx.x; i < D; i += blockDim.x)
        orow[i] = (_Float16)(row[i] * scale * g[i]);
}

__global__ __launch_bounds__(256)
void rmsnorm_silu_f32(const float* __restrict__ x, const float* __restrict__ g,
                      float* __restrict__ out, int D) {
    const float* row = x + (size_t)blockIdx.x * D;
    float tot = block_sumsq(row, D);
    float scale = rsqrtf(tot / (float)D + 1e-5f);
    float* orow = out + (size_t)blockIdx.x * D;
    for (int i = threadIdx.x; i < D; i += blockDim.x) {
        float v = row[i] * scale * g[i];
        orow[i] = siluf_(v);
    }
}

// ---------------------------------------------------------------------------
// causal depthwise conv (D_CONV=4) + bias + silu over xs (= xz[..., :DI])
// writes f32 (for scan) and f16 (for x_proj GEMM)
// ---------------------------------------------------------------------------
__global__ __launch_bounds__(256)
void conv_silu_kernel(const float* __restrict__ xz, const float* __restrict__ cw,
                      const float* __restrict__ cb,
                      float* __restrict__ xc32, _Float16* __restrict__ xc16) {
    long idx = (long)blockIdx.x * blockDim.x + threadIdx.x;
    const long total = (long)MTOK * DI;
    if (idx >= total) return;
    int  d  = (int)(idx & (DI - 1));
    long bt = idx >> 10;
    int  t  = (int)(bt & (TT - 1));
    long b  = bt >> 10;
    float acc = cb[d];
#pragma unroll
    for (int k = 0; k < 4; ++k) {
        int tt = t - 3 + k;
        if (tt >= 0)
            acc += cw[d * 4 + k] * xz[((b << 10) + tt) * (long)(2 * DI) + d];
    }
    float v = siluf_(acc);
    xc32[idx] = v;
    xc16[idx] = (_Float16)v;
}

// ---------------------------------------------------------------------------
// selective scan: one thread per (b, d); h[16] in registers; T=1024 steps
// y16 = (scan_y + xc*Dp) * silu(z)  (f16, ready for out_proj GEMM)
// ---------------------------------------------------------------------------
__global__ __launch_bounds__(256)
void scan_kernel(const float* __restrict__ dt, const float* __restrict__ xc,
                 const float* __restrict__ dbl, const float* __restrict__ xz,
                 const float* __restrict__ A_log, const float* __restrict__ D_p,
                 _Float16* __restrict__ y16) {
    const int d = blockIdx.x * blockDim.x + threadIdx.x;   // 0..DI-1
    const int b = blockIdx.y;                              // 0..BB-1
    float A[DSZ];
#pragma unroll
    for (int s = 0; s < DSZ; ++s) A[s] = -__expf(A_log[(size_t)d * DSZ + s]);
    const float Dp = D_p[d];
    float h[DSZ];
#pragma unroll
    for (int s = 0; s < DSZ; ++s) h[s] = 0.0f;

    for (int t = 0; t < TT; ++t) {
        const long base = (long)(b * TT + t);
        const float dtv = dt[base * DI + d];
        const float xv  = xc[base * DI + d];
        const float zv  = xz[base * (long)(2 * DI) + DI + d];
        const float4* Bv4 = reinterpret_cast<const float4*>(dbl + base * 64 + DTRK);
        const float4* Cv4 = reinterpret_cast<const float4*>(dbl + base * 64 + DTRK + DSZ);
        float Bv[DSZ], Cv[DSZ];
#pragma unroll
        for (int q = 0; q < 4; ++q) {
            float4 bq = Bv4[q]; float4 cq = Cv4[q];
            Bv[4*q+0]=bq.x; Bv[4*q+1]=bq.y; Bv[4*q+2]=bq.z; Bv[4*q+3]=bq.w;
            Cv[4*q+0]=cq.x; Cv[4*q+1]=cq.y; Cv[4*q+2]=cq.z; Cv[4*q+3]=cq.w;
        }
        const float dx = dtv * xv;
        float y = 0.0f;
#pragma unroll
        for (int s = 0; s < DSZ; ++s) {
            float dA = __expf(dtv * A[s]);
            h[s] = dA * h[s] + dx * Bv[s];
            y   += h[s] * Cv[s];
        }
        y += xv * Dp;
        y16[base * DI + d] = (_Float16)(y * siluf_(zv));
    }
}

// ---------------------------------------------------------------------------
// host side
// ---------------------------------------------------------------------------
static inline int cdivl(long a, long b) { return (int)((a + b - 1) / b); }

extern "C" void kernel_launch(void* const* d_in, const int* in_sizes, int n_in,
                              void* d_out, int out_size, void* d_ws, size_t ws_size,
                              hipStream_t stream) {
    const float* state   = (const float*)d_in[0];
    const float* pa      = (const float*)d_in[1];
    const float* state_w = (const float*)d_in[2];
    const float* state_b = (const float*)d_in[3];
    const float* act_w   = (const float*)d_in[4];
    const float* act_b   = (const float*)d_in[5];
    const float* fuse_w  = (const float*)d_in[6];
    const float* fuse_b  = (const float*)d_in[7];
    const float* fuse_g  = (const float*)d_in[8];
    const float* norm_g  = (const float*)d_in[9];
    const float* in_w    = (const float*)d_in[10];
    const float* conv_w  = (const float*)d_in[11];
    const float* conv_b  = (const float*)d_in[12];
    const float* xproj_w = (const float*)d_in[13];
    const float* dt_w    = (const float*)d_in[14];
    const float* dt_b    = (const float*)d_in[15];
    const float* A_log   = (const float*)d_in[16];
    const float* D_p     = (const float*)d_in[17];
    const float* out_w   = (const float*)d_in[18];
    const float* head_w  = (const float*)d_in[19];
    const float* head_b  = (const float*)d_in[20];
    float* out = (float*)d_out;

    // ---- workspace arena ----
    char*  ws  = (char*)d_ws;
    size_t off = 0;
    auto carve = [&](size_t bytes) -> void* {
        off = (off + 255) & ~(size_t)255;
        void* p = ws + off;
        off += bytes;
        return p;
    };
    const long M = MTOK;

    _Float16* fw16  = (_Float16*)carve((size_t)DM * 640 * 2);
    _Float16* inw16 = (_Float16*)carve((size_t)NL * 2 * DI * DM * 2);
    _Float16* xpw16 = (_Float16*)carve((size_t)NL * 64 * DI * 2);
    _Float16* dtw16 = (_Float16*)carve((size_t)NL * DI * DTRK * 2);
    _Float16* ow16  = (_Float16*)carve((size_t)NL * DM * DI * 2);
    _Float16* hw16  = (_Float16*)carve((size_t)32 * DM * 2);

    float*    xbuf  = (float*)   carve((size_t)M * DM * 4);
    _Float16* nrm16 = (_Float16*)carve((size_t)M * DM * 2);
    char*     xzraw = (char*)    carve((size_t)M * 2 * DI * 4);
    float*    xz    = (float*)xzraw;
    float*    xc32  = (float*)   carve((size_t)M * DI * 4);
    _Float16* xc16  = (_Float16*)carve((size_t)M * DI * 2);
    float*    dbl   = (float*)   carve((size_t)M * 64 * 4);
    _Float16* dtr16 = (_Float16*)carve((size_t)M * DTRK * 2);
    float*    dt32  = (float*)   carve((size_t)M * DI * 4);
    _Float16* y16   = (_Float16*)carve((size_t)M * DI * 2);

    // front-end buffers aliased into xz (disjoint lifetimes):
    _Float16* fin16  = (_Float16*)xzraw;                     // 16384*640 f16 = 21 MB
    float*    fused  = (float*)(xzraw + ((size_t)32 << 20)); // at +32MB, 33.5 MB
    _Float16* x16    = nrm16;                                // head staging reuses nrm16

    (void)n_in; (void)in_sizes; (void)out_size; (void)ws_size;

    // ---- weight conversion to f16 (K padded to multiple of 32) ----
    {
        long t;
        t = (long)DM * 640;
        f32_to_f16_pad<<<cdivl(t,256),256,0,stream>>>(fuse_w,  fw16, t, 640, 640);
        t = (long)NL * 2 * DI * DM;
        f32_to_f16_pad<<<cdivl(t,256),256,0,stream>>>(in_w,    inw16, t, DM, DM);
        t = (long)NL * 64 * DI;
        f32_to_f16_pad<<<cdivl(t,256),256,0,stream>>>(xproj_w, xpw16, t, DI, DI);
        t = (long)NL * DI * DTRK;
        f32_to_f16_pad<<<cdivl(t,256),256,0,stream>>>(dt_w,    dtw16, t, DTRK, DTRK);
        t = (long)NL * DM * DI;
        f32_to_f16_pad<<<cdivl(t,256),256,0,stream>>>(out_w,   ow16,  t, DI, DI);
        t = (long)32 * DM;
        f32_to_f16_pad<<<cdivl(t,256),256,0,stream>>>(head_w,  hw16,  t, DM, DM);
    }

    const dim3 blk(256);
    const int  gM = (int)(M / 128);   // 128-row block tiles in M (both configs)

    // big-N config: wave tile 64x64, block tile 128x256 (N % 256 workloads)
    auto gemm_big = [&](const _Float16* Aa, const _Float16* Ww, float* Cc,
                        int N, int K, const float* bias, int mode) {
        gemm_f16_wmma<4,4,2,4><<<dim3(gM, cdivl(N,256)), blk, 0, stream>>>(
            Aa, Ww, Cc, (int)M, N, K, bias, mode);
    };
    // small-N config: wave tile 32x32, block tile 128x64
    auto gemm_small = [&](const _Float16* Aa, const _Float16* Ww, float* Cc,
                          int N, int K, const float* bias, int mode) {
        gemm_f16_wmma<2,2,4,2><<<dim3(gM, cdivl(N,64)), blk, 0, stream>>>(
            Aa, Ww, Cc, (int)M, N, K, bias, mode);
    };

    // ---- front end ----
    {
        long t = M * (DM + 128);
        embed_kernel<<<cdivl(t,256),blk,0,stream>>>(state, pa, state_w, state_b, act_w, act_b, fin16);
        gemm_big(fin16, fw16, fused, DM, 640, fuse_b, EPI_BIAS);
        rmsnorm_silu_f32<<<(int)M,blk,0,stream>>>(fused, fuse_g, xbuf, DM);
    }

    // ---- mamba layers ----
    for (int L = 0; L < NL; ++L) {
        rmsnorm_to_f16<<<(int)M,blk,0,stream>>>(xbuf, norm_g + (size_t)L*DM, nrm16, DM);

        gemm_big(nrm16, inw16 + (size_t)L*2*DI*DM, xz, 2*DI, DM, nullptr, EPI_NONE);

        conv_silu_kernel<<<cdivl(M*DI,256),blk,0,stream>>>(
            xz, conv_w + (size_t)L*DI*4, conv_b + (size_t)L*DI, xc32, xc16);

        gemm_small(xc16, xpw16 + (size_t)L*64*DI, dbl, 64, DI, nullptr, EPI_NONE);

        f32_to_f16_pad<<<cdivl(M*DTRK,256),blk,0,stream>>>(dbl, dtr16, M*DTRK, 64, DTRK);

        gemm_big(dtr16, dtw16 + (size_t)L*DI*DTRK, dt32, DI, DTRK,
                 dt_b + (size_t)L*DI, EPI_BIAS_SOFTPLUS);

        scan_kernel<<<dim3(DI/256, BB),blk,0,stream>>>(
            dt32, xc32, dbl, xz, A_log + (size_t)L*DI*DSZ, D_p + (size_t)L*DI, y16);

        gemm_big(y16, ow16 + (size_t)L*DM*DI, xbuf, DM, DI, nullptr, EPI_ACC);
    }

    // ---- head ----
    f32_to_f16_pad<<<cdivl(M*DM,256),blk,0,stream>>>(xbuf, x16, M*DM, DM, DM);
    gemm_small(x16, hw16, out, 32, DM, head_b, EPI_BIAS);
}
